// TreeAttention_51565377356140
// MI455X (gfx1250) — matrix operand order repaired and checked by reading, not compiled, not run on hardware
//
#include <hip/hip_runtime.h>
#include <math.h>

// ---------------- problem constants ----------------
#define ROWS    8192        // B*S
#define DIN     1024
#define DOUT    1024
#define NTREES  12
#define NINT    7
#define NLEAF   8
#define NHEADS  3
#define NC1     96          // 84 decision cols + 12 gate cols
#define KSTEPS1 32          // DIN / 32
#define FRAG_SHORTS 512     // 32 lanes * 16 bf16 per WMMA fragment
#define B1_FRAGS (6*KSTEPS1)      // 192  (6 n-tiles x 32 k-steps)
#define B2_FRAGS (NHEADS*64*3)    // 576  (3 heads x 64 n-tiles x 3 k-steps)

typedef __attribute__((ext_vector_type(16))) __bf16        bf16x16;
typedef __attribute__((ext_vector_type(8)))  float         f32x8;
typedef __attribute__((ext_vector_type(4)))  float         f32x4;
typedef __attribute__((ext_vector_type(4)))  unsigned int  u32x4;

union Frag16 {
  bf16x16        v;
  unsigned short s[16];
  u32x4          q[2];
};

__device__ __forceinline__ unsigned short f2bf(float f) {
  union { float f; unsigned u; } cv; cv.f = f;
  unsigned r = cv.u + 0x7fffu + ((cv.u >> 16) & 1u);  // round-to-nearest-even
  return (unsigned short)(r >> 16);
}

// ---------------------------------------------------------------------------
// Kernel 0: pack decision+gate weights and leaf outputs into bf16 WMMA
// B-fragment layout. B 32x16 layout: lane n<16 holds column n; VGPR j holds
// K = 2j, 2j+1 (j<4) or 2j+8, 2j+9 (j>=4); lanes 16-31 hold the same columns with K offset +8.
// => lane element e maps to k_local = 8*half + e (e<8), 16 + 8*half + (e-8) (e>=8).
// ---------------------------------------------------------------------------
__global__ __launch_bounds__(256) void tree_pack_weights(
    const float* __restrict__ dw,   // [12,7,1024]
    const float* __restrict__ gw,   // [12,1024]
    const float* __restrict__ lo,   // [3,12,8,1024]
    unsigned short* __restrict__ wsB1,
    unsigned short* __restrict__ wsB2)
{
  const int t    = blockIdx.x * 256 + threadIdx.x;
  const int lane = t & 31;
  const int frag = t >> 5;
  const int nn   = lane & 15;
  const int half = lane >> 4;

  if (frag < B1_FRAGS) {
    const int nTile = frag / KSTEPS1;
    const int ks    = frag % KSTEPS1;
    const int c     = nTile * 16 + nn;             // combined output column
    unsigned short* dst = wsB1 + (size_t)frag * FRAG_SHORTS + lane * 16;
    const int kBase = ks * 32;
#pragma unroll
    for (int e = 0; e < 16; ++e) {
      const int k = kBase + ((e < 8) ? (8 * half + e) : (16 + 8 * half + (e - 8)));
      const float v = (c < 84) ? dw[(size_t)c * DIN + k]
                               : gw[(size_t)(c - 84) * DIN + k];
      dst[e] = f2bf(v);
    }
  } else if (frag < B1_FRAGS + B2_FRAGS) {
    const int g     = frag - B1_FRAGS;
    const int h     = g / 192;
    const int r     = g % 192;
    const int nTile = r / 3;
    const int ks    = r % 3;
    const int col   = nTile * 16 + nn;
    unsigned short* dst = wsB2 + (size_t)g * FRAG_SHORTS + lane * 16;
#pragma unroll
    for (int e = 0; e < 16; ++e) {
      const int k    = ks * 32 + ((e < 8) ? (8 * half + e) : (16 + 8 * half + (e - 8)));    // 0..95 = t*8 + leaf
      const int tr   = k >> 3;
      const int leaf = k & 7;
      dst[e] = f2bf(lo[(((size_t)h * NTREES + tr) * NLEAF + leaf) * DOUT + col]);
    }
  }
}

// ---------------------------------------------------------------------------
// Kernel 1: routing GEMM (x[16,1024] x W^T[1024,96]) via bf16 WMMA, then
// sigmoid/temperature, root->leaf path products, tree softmax; emits routed
// probabilities pre-packed as phase-2 A fragments (bf16).
// A 16x32 bf16 layout: lane row m = lane&15; VGPR0-3 = K 0..7 (+8 if half),
// VGPR4-7 = K 16..23 (+8 if half) -> two contiguous 8-element runs.
// ---------------------------------------------------------------------------
__global__ __launch_bounds__(128) void tree_phase1(
    const float* __restrict__ x,      // [8192,1024]
    const float* __restrict__ dbias,  // [12,7]
    const float* __restrict__ ntl,    // [12,7]
    const float* __restrict__ gbias,  // [12]
    const unsigned short* __restrict__ wsB1,
    unsigned short* __restrict__ wsA2)
{
  __shared__ float s_logits[4][16 * NC1];
  __shared__ float s_routed[4][16 * NC1];
  __shared__ float s_invtemp[84];
  __shared__ float s_bias[NC1];

  const int tid  = threadIdx.x;
  const int wave = tid >> 5;
  const int lane = tid & 31;
  const int n    = lane & 15;
  const int half = lane >> 4;

  if (tid < 84) s_invtemp[tid] = 1.0f / log1pf(expf(ntl[tid] + 0.5413f));
  if (tid < NC1) s_bias[tid] = (tid < 84) ? dbias[tid] : gbias[tid - 84];
  __syncthreads();

  const int rowTile = blockIdx.x * 4 + wave;
  const int rowBase = rowTile * 16;
  const float* xr = x + (size_t)(rowBase + n) * DIN;

  const f32x8 zero8 = {0.f, 0.f, 0.f, 0.f, 0.f, 0.f, 0.f, 0.f};
  f32x8 acc[6];
#pragma unroll
  for (int i = 0; i < 6; ++i) acc[i] = zero8;

  for (int ks = 0; ks < KSTEPS1; ++ks) {
    // A fragment straight from global: two contiguous 8-float runs per lane.
    Frag16 a;
    const f32x4* p0 = (const f32x4*)(xr + ks * 32 + 8 * half);
    const f32x4* p1 = (const f32x4*)(xr + ks * 32 + 16 + 8 * half);
    const f32x4 r0 = p0[0], r1 = p0[1], r2 = p1[0], r3 = p1[1];
#pragma unroll
    for (int i = 0; i < 4; ++i) {
      a.s[i]      = f2bf(r0[i]);
      a.s[4 + i]  = f2bf(r1[i]);
      a.s[8 + i]  = f2bf(r2[i]);
      a.s[12 + i] = f2bf(r3[i]);
    }
#pragma unroll
    for (int nt = 0; nt < 6; ++nt) {
      Frag16 b;
      const u32x4* bp = (const u32x4*)(wsB1 +
          ((size_t)(nt * KSTEPS1 + ks) * 32 + lane) * 16);
      b.q[0] = bp[0]; b.q[1] = bp[1];
      acc[nt] = __builtin_amdgcn_wmma_f32_16x16x32_bf16(
          false, a.v, false, b.v, (short)0, acc[nt], false, false);
    }
  }

  // C layout: lane holds col n of n-tile; VGPR v = row v (+8 if half).
#pragma unroll
  for (int nt = 0; nt < 6; ++nt) {
#pragma unroll
    for (int v = 0; v < 8; ++v) {
      const int m = v + 8 * half;
      const int c = nt * 16 + n;
      s_logits[wave][m * NC1 + c] = acc[nt][v] + s_bias[c];
    }
  }
  __syncthreads();

  // Elementwise: 16 rows x 12 trees = 192 (row,tree) pairs per wave.
  const float* lg = s_logits[wave];
  float*       rt = s_routed[wave];
#pragma unroll
  for (int it = 0; it < 6; ++it) {
    const int idx = it * 32 + lane;
    const int row = idx / 12;
    const int tr  = idx % 12;
    float d[7];
#pragma unroll
    for (int nd = 0; nd < 7; ++nd) {
      const float z = lg[row * NC1 + tr * 7 + nd] * s_invtemp[tr * 7 + nd];
      d[nd] = 1.0f / (1.0f + expf(-z));
    }
    float gmax = -3.4e38f;
#pragma unroll
    for (int tt = 0; tt < 12; ++tt) gmax = fmaxf(gmax, lg[row * NC1 + 84 + tt]);
    float gsum = 0.f;
#pragma unroll
    for (int tt = 0; tt < 12; ++tt) gsum += expf(lg[row * NC1 + 84 + tt] - gmax);
    const float w = expf(lg[row * NC1 + 84 + tr] - gmax) / gsum;
#pragma unroll
    for (int leaf = 0; leaf < 8; ++leaf) {
      const int b2 = (leaf >> 2) & 1, b1 = (leaf >> 1) & 1, b0 = leaf & 1;
      const float p0 = b2 ? (1.f - d[0]) : d[0];
      const float p1 = b1 ? (1.f - d[1 + b2]) : d[1 + b2];
      const float p2 = b0 ? (1.f - d[3 + 2 * b2 + b1]) : d[3 + 2 * b2 + b1];
      rt[row * NC1 + tr * 8 + leaf] = w * p0 * p1 * p2;
    }
  }
  __syncthreads();

  // Pack routed probs as phase-2 A fragments (bf16, A 16x32 layout).
#pragma unroll
  for (int ks = 0; ks < 3; ++ks) {
    Frag16 fr;
#pragma unroll
    for (int e = 0; e < 16; ++e) {
      const int kl = (e < 8) ? (8 * half + e) : (16 + 8 * half + (e - 8));
      fr.s[e] = f2bf(rt[n * NC1 + ks * 32 + kl]);
    }
    u32x4* dst = (u32x4*)(wsA2 + ((size_t)(rowTile * 3 + ks) * 32 + lane) * 16);
    dst[0] = fr.q[0]; dst[1] = fr.q[1];
  }
}

// ---------------------------------------------------------------------------
// Kernel 2: out[h, row, col] = routed[row, :96] . leafmat[h, :96, col]
// K=96 -> 3 WMMAs per 16x16 tile; 8 n-tiles per wave to amortize A loads.
// Store-bandwidth bound: 96 MB fp32 output.
// ---------------------------------------------------------------------------
__global__ __launch_bounds__(128) void tree_phase2(
    const unsigned short* __restrict__ wsA2,
    const unsigned short* __restrict__ wsB2,
    float* __restrict__ out)
{
  const int tid  = threadIdx.x;
  const int lane = tid & 31;
  const int n    = lane & 15;
  const int half = lane >> 4;
  const int wg   = blockIdx.x * 4 + (tid >> 5);   // 0..12287
  const int h       = wg / 4096;                  // 512 rowTiles * 8 groups
  const int rem     = wg % 4096;
  const int rowTile = rem >> 3;
  const int ngrp    = rem & 7;

  Frag16 a[3];
#pragma unroll
  for (int ks = 0; ks < 3; ++ks) {
    const u32x4* ap = (const u32x4*)(wsA2 +
        ((size_t)(rowTile * 3 + ks) * 32 + lane) * 16);
    a[ks].q[0] = ap[0]; a[ks].q[1] = ap[1];
  }

  const int rowBase = rowTile * 16;
  const f32x8 zero8 = {0.f, 0.f, 0.f, 0.f, 0.f, 0.f, 0.f, 0.f};
#pragma unroll
  for (int nt = 0; nt < 8; ++nt) {
    const int nTile = ngrp * 8 + nt;
    f32x8 acc = zero8;
#pragma unroll
    for (int ks = 0; ks < 3; ++ks) {
      Frag16 b;
      const u32x4* bp = (const u32x4*)(wsB2 +
          ((size_t)((h * 64 + nTile) * 3 + ks) * 32 + lane) * 16);
      b.q[0] = bp[0]; b.q[1] = bp[1];
      acc = __builtin_amdgcn_wmma_f32_16x16x32_bf16(
          false, a[ks].v, false, b.v, (short)0, acc, false, false);
    }
    float* op = out + ((size_t)h * ROWS + rowBase) * DOUT + nTile * 16 + n;
#pragma unroll
    for (int v = 0; v < 8; ++v)
      op[(size_t)(v + 8 * half) * DOUT] = acc[v];
  }
}

// ---------------------------------------------------------------------------
extern "C" void kernel_launch(void* const* d_in, const int* in_sizes, int n_in,
                              void* d_out, int out_size, void* d_ws, size_t ws_size,
                              hipStream_t stream) {
  (void)in_sizes; (void)n_in; (void)out_size; (void)ws_size;
  const float* x   = (const float*)d_in[0];   // [4,2048,1024]
  const float* dw  = (const float*)d_in[1];   // [12,7,1024]
  const float* db  = (const float*)d_in[2];   // [12,7]
  const float* ntl = (const float*)d_in[3];   // [12,7]
  const float* gw  = (const float*)d_in[4];   // [12,1024]
  const float* gb  = (const float*)d_in[5];   // [12]
  const float* lo  = (const float*)d_in[6];   // [3,12,8,1024]
  float* out = (float*)d_out;                 // [3,4,2048,1024]

  unsigned short* wsB1 = (unsigned short*)d_ws;
  unsigned short* wsB2 = wsB1 + (size_t)B1_FRAGS * FRAG_SHORTS;                 // +192 frags
  unsigned short* wsA2 = wsB1 + (size_t)(B1_FRAGS + B2_FRAGS) * FRAG_SHORTS;    // +768 frags

  // (192 + 576) fragments * 32 lanes = 24576 threads
  tree_pack_weights<<<96, 256, 0, stream>>>(dw, gw, lo, wsB1, wsB2);
  // 8192 rows / (4 waves * 16 rows) = 128 blocks
  tree_phase1<<<128, 128, 0, stream>>>(x, db, ntl, gb, wsB1, wsA2);
  // 3 heads * 512 rowTiles * 8 col-groups = 12288 waves / 4 = 3072 blocks
  tree_phase2<<<3072, 128, 0, stream>>>(wsA2, wsB2, out);
}
